// PNANet_46746424049890
// MI455X (gfx1250) — matrix operand order repaired and checked by reading, not compiled, not run on hardware
//
#include <hip/hip_runtime.h>
#include <math.h>
#include <stdint.h>

// ---------------------------------------------------------------------------
// CDNA5 / gfx1250 PNA forward. wave32. WMMA bf16 16x16x32 for all GEMMs.
// bf16 operands pre-converted once; GEMM stages A via async global->LDS
// (ASYNCcnt) with double-buffered LDS, B via transposed register staging.
// ---------------------------------------------------------------------------

typedef __bf16  v16bf __attribute__((ext_vector_type(16)));
typedef __bf16  v8bf  __attribute__((ext_vector_type(8)));
typedef __bf16  v4bf  __attribute__((ext_vector_type(4)));
typedef __bf16  v2bf  __attribute__((ext_vector_type(2)));
typedef float   v8f   __attribute__((ext_vector_type(8)));
typedef float   v4f   __attribute__((ext_vector_type(4)));

// async builtin takes int4 pointers in explicit address spaces
typedef int v4i __attribute__((vector_size(16)));
typedef __attribute__((address_space(1))) v4i* gv4i_p;   // global
typedef __attribute__((address_space(3))) v4i* lv4i_p;   // LDS

#if __has_builtin(__builtin_amdgcn_global_load_async_to_lds_b128)
#define HAVE_ASYNC_LDS 1
#else
#define HAVE_ASYNC_LDS 0
#endif

__device__ __forceinline__ v4bf cvt4(v4f a) {
    v4bf r;
    r[0] = (__bf16)a[0]; r[1] = (__bf16)a[1];
    r[2] = (__bf16)a[2]; r[3] = (__bf16)a[3];
    return r;
}

// ------------------------------- small utils -------------------------------

__global__ __launch_bounds__(256) void k_zero_i32(int* __restrict__ p, int n) {
    int i = blockIdx.x * 256 + threadIdx.x;
    if (i < n) p[i] = 0;
}

__global__ __launch_bounds__(256) void k_f32_to_bf16(const float* __restrict__ in,
                                                     __bf16* __restrict__ out, int n4) {
    int i = blockIdx.x * 256 + threadIdx.x;
    if (i < n4) *(v4bf*)(out + i * 4) = cvt4(*(const v4f*)(in + i * 4));
}

__global__ __launch_bounds__(256) void k_degree(const long long* __restrict__ ei,
                                                int* __restrict__ deg, int E) {
    int e = blockIdx.x * 256 + threadIdx.x;
    if (e < E) atomicAdd(&deg[(int)ei[(size_t)E + e]], 1);
}

__global__ __launch_bounds__(256) void k_scan_amp(const int* __restrict__ deg,
                                                  int* __restrict__ rowptr,
                                                  float* __restrict__ amp,
                                                  float* __restrict__ inva, int N) {
    __shared__ float red[256];
    __shared__ float s_avg;
    int t = threadIdx.x;
    float ls = 0.f;
    for (int n = t; n < N; n += 256) ls += logf((float)deg[n] + 1.f);
    red[t] = ls; __syncthreads();
    for (int s = 128; s > 0; s >>= 1) { if (t < s) red[t] += red[t + s]; __syncthreads(); }
    if (t == 0) {
        s_avg = red[0] / (float)N;
        int acc = 0;
        for (int n = 0; n < N; n++) { rowptr[n] = acc; acc += deg[n]; }
        rowptr[N] = acc;
    }
    __syncthreads();
    float avg = s_avg;
    for (int n = t; n < N; n += 256) {
        float a = logf(fmaxf((float)deg[n], 1.f) + 1.f) / avg;
        amp[n] = a; inva[n] = 1.f / a;
    }
}

__global__ __launch_bounds__(256) void k_fill_csr(const long long* __restrict__ ei,
                                                  const int* __restrict__ rowptr,
                                                  int* __restrict__ cursor,
                                                  int* __restrict__ ebuf, int E) {
    int e = blockIdx.x * 256 + threadIdx.x;
    if (e >= E) return;
    int src = (int)ei[e];
    int dst = (int)ei[(size_t)E + e];
    int pos = atomicAdd(&cursor[dst], 1);
    ebuf[rowptr[dst] + pos] = src;
}

__global__ __launch_bounds__(256) void k_sort_buckets(const int* __restrict__ rowptr,
                                                      int* __restrict__ ebuf, int N) {
    int n = blockIdx.x * 256 + threadIdx.x;
    if (n >= N) return;
    int s = rowptr[n], e = rowptr[n + 1];
    for (int i = s + 1; i < e; i++) {
        int v = ebuf[i]; int j = i - 1;
        while (j >= s && ebuf[j] > v) { ebuf[j + 1] = ebuf[j]; j--; }
        ebuf[j + 1] = v;
    }
}

// ------------------------------- WMMA GEMM ---------------------------------
// C[M,N](f32) = A[M,K](bf16) * B[K,N](bf16). Block 256 thr, tile 64x128,
// BK=32; wave (wm,wn) owns 32x32 = 2x2 WMMA tiles. A tile: async global->LDS
// (double-buffered). B tile: register-prefetched, stored transposed [n][k].
template <bool ACCUM, bool RELU, bool BIAS, bool SCALE>
__global__ __launch_bounds__(256) void k_gemm_bf16(
    const __bf16* __restrict__ A, const __bf16* __restrict__ B, float* __restrict__ C,
    int K, int lda, int ldb, int ldc,
    const float* __restrict__ bias, const float* __restrict__ rowscale) {

    __shared__ __bf16 As[2][64][32];   // [buf][m][k]  8 KB
    __shared__ __bf16 Bs[128][32];     // [n][k]       8 KB

    const int t    = threadIdx.x;
    const int lane = t & 31;
    const int w    = t >> 5;
    const int wm   = w >> 2;             // 0..1
    const int wn   = w & 3;              // 0..3
    const int row0 = blockIdx.y * 64;
    const int col0 = blockIdx.x * 128;

    // A staging: wave w copies rows w*8..w*8+7; lane -> 16B segment
    const int arw = w * 8 + (lane >> 2);
    const int acw = (lane & 3) * 8;
    // B staging: thread t -> k-pair (t>>4)*2, n-octet (t&15)*8
    const int bk = (t >> 4) * 2;
    const int bn = (t & 15) * 8;

    v8bf rB0, rB1;

    auto gloadB = [&](int k0) {
        rB0 = *(const v8bf*)(B + (k0 + bk) * ldb + col0 + bn);
        rB1 = *(const v8bf*)(B + (k0 + bk + 1) * ldb + col0 + bn);
    };
    auto storeB = [&]() {
#pragma unroll
        for (int j = 0; j < 8; j++) {
            v2bf p; p[0] = rB0[j]; p[1] = rB1[j];
            *(v2bf*)&Bs[bn + j][bk] = p;       // ds_store_b32
        }
    };
    auto stageA = [&](int k0, int buf) {
        const __bf16* gp = A + (row0 + arw) * lda + k0 + acw;
        __bf16* lp = &As[buf][arw][acw];
#if HAVE_ASYNC_LDS
        __builtin_amdgcn_global_load_async_to_lds_b128((gv4i_p)gp, (lv4i_p)lp, 0, 0);
#else
        *(v8bf*)lp = *(const v8bf*)gp;         // global b128 -> ds b128
#endif
    };

    const int m = lane & 15;
    const int g = lane >> 4;

    v8f acc[2][2] = {};
    int buf = 0;

    stageA(0, 0);
    gloadB(0);
    for (int k0 = 0; k0 < K; k0 += 32) {
        __syncthreads();                       // prev compute done
        storeB();
        const bool more = (k0 + 32 < K);
        if (more) gloadB(k0 + 32);
#if HAVE_ASYNC_LDS
        asm volatile("s_wait_asynccnt 0x0" ::: "memory");
#endif
        __syncthreads();                       // tiles visible to all waves
        if (more) stageA(k0 + 32, buf ^ 1);    // overlaps with WMMAs below

        v16bf afr[2], bfr[2];
#pragma unroll
        for (int mi = 0; mi < 2; mi++) {
            const __bf16* ap = &As[buf][wm * 32 + mi * 16 + m][0];
            v8bf lo = *(const v8bf*)(ap + g * 8);          // ds_load_b128
            v8bf hi = *(const v8bf*)(ap + 16 + g * 8);
            afr[mi] = __builtin_shufflevector(lo, hi,
                0, 1, 2, 3, 4, 5, 6, 7, 8, 9, 10, 11, 12, 13, 14, 15);
        }
#pragma unroll
        for (int ni = 0; ni < 2; ni++) {
            const __bf16* bp = &Bs[wn * 32 + ni * 16 + m][0];
            v8bf lo = *(const v8bf*)(bp + g * 8);
            v8bf hi = *(const v8bf*)(bp + 16 + g * 8);
            bfr[ni] = __builtin_shufflevector(lo, hi,
                0, 1, 2, 3, 4, 5, 6, 7, 8, 9, 10, 11, 12, 13, 14, 15);
        }
#pragma unroll
        for (int mi = 0; mi < 2; mi++)
#pragma unroll
            for (int ni = 0; ni < 2; ni++)
                acc[mi][ni] = __builtin_amdgcn_wmma_f32_16x16x32_bf16(
                    false, afr[mi], false, bfr[ni],
                    (short)0, acc[mi][ni], false, false);
        buf ^= 1;
    }

#pragma unroll
    for (int mi = 0; mi < 2; mi++) {
#pragma unroll
        for (int ni = 0; ni < 2; ni++) {
            const int colC = col0 + wn * 32 + ni * 16 + m;
            const int rowB = row0 + wm * 32 + mi * 16 + g * 8;
            const float bv = BIAS ? bias[colC] : 0.f;
#pragma unroll
            for (int r = 0; r < 8; r++) {
                int rr = rowB + r;
                float v = acc[mi][ni][r];
                if (SCALE) v *= rowscale[rr];
                float* cp = C + rr * ldc + colC;
                if (ACCUM) v += *cp;
                if (BIAS)  v += bv;
                if (RELU)  v = fmaxf(v, 0.f);
                *cp = v;
            }
        }
    }
}

// -------------------------- fused edge aggregation --------------------------
// layer 1: f32 in (A1/B1), bf16 out (feeds WMMA GEMMs). One block per node.
template <int NV4>
__global__ __launch_bounds__(256) void k_edge_aggr_v(
    const float* __restrict__ Ad, const float* __restrict__ Bsrc,
    const int* __restrict__ rowptr, const int* __restrict__ ebuf,
    __bf16* __restrict__ meanO, __bf16* __restrict__ mnO,
    __bf16* __restrict__ mxO, __bf16* __restrict__ stdO, int F) {

    const int n = blockIdx.x, t = threadIdx.x;
    const int start = rowptr[n], end = rowptr[n + 1];
    const int deg = end - start;
    const float safe = (float)(deg > 0 ? deg : 1);
    const int base = n * F + t * 4;

    v4f an[NV4], s[NV4], s2[NV4], mnv[NV4], mxv[NV4];
#pragma unroll
    for (int j = 0; j < NV4; j++) {
        an[j] = *(const v4f*)(Ad + base + j * 1024);
#pragma unroll
        for (int c = 0; c < 4; c++) {
            s[j][c] = 0.f; s2[j][c] = 0.f;
            mnv[j][c] = INFINITY; mxv[j][c] = -INFINITY;
        }
    }
    for (int e = start; e < end; e++) {
        int src = ebuf[e];
        if (e + 1 < end) __builtin_prefetch(&Bsrc[ebuf[e + 1] * F + t * 4], 0, 0);
        const float* br = Bsrc + src * F + t * 4;
#pragma unroll
        for (int j = 0; j < NV4; j++) {
            v4f b = *(const v4f*)(br + j * 1024);
#pragma unroll
            for (int c = 0; c < 4; c++) {
                float mv = an[j][c] + b[c];
                s[j][c] += mv; s2[j][c] += mv * mv;
                mnv[j][c] = fminf(mnv[j][c], mv);
                mxv[j][c] = fmaxf(mxv[j][c], mv);
            }
        }
    }
#pragma unroll
    for (int j = 0; j < NV4; j++) {
        v4f me, sd, lo, hi;
#pragma unroll
        for (int c = 0; c < 4; c++) {
            float mean = s[j][c] / safe;
            float var  = fmaxf(s2[j][c] / safe - mean * mean, 0.f);
            me[c] = mean;
            sd[c] = sqrtf(var + 1e-5f);
            lo[c] = deg > 0 ? mnv[j][c] : 0.f;
            hi[c] = deg > 0 ? mxv[j][c] : 0.f;
        }
        *(v4bf*)(meanO + base + j * 1024) = cvt4(me);
        *(v4bf*)(mnO   + base + j * 1024) = cvt4(lo);
        *(v4bf*)(mxO   + base + j * 1024) = cvt4(hi);
        *(v4bf*)(stdO  + base + j * 1024) = cvt4(sd);
    }
}

// layer 2: f32 in/out (feeds scalar post stage), F == 256
__global__ __launch_bounds__(256) void k_edge_aggr_s(
    const float* __restrict__ Ad, const float* __restrict__ Bsrc,
    const int* __restrict__ rowptr, const int* __restrict__ ebuf,
    float* __restrict__ meanO, float* __restrict__ mnO,
    float* __restrict__ mxO, float* __restrict__ stdO, int F) {

    const int n = blockIdx.x, t = threadIdx.x;
    const int start = rowptr[n], end = rowptr[n + 1];
    const int deg = end - start;
    const float safe = (float)(deg > 0 ? deg : 1);
    const int o = n * F + t;

    float an = Ad[o];
    float s = 0.f, s2 = 0.f, mnv = INFINITY, mxv = -INFINITY;
    for (int e = start; e < end; e++) {
        int src = ebuf[e];
        if (e + 1 < end) __builtin_prefetch(&Bsrc[ebuf[e + 1] * F + t], 0, 0);
        float mv = an + Bsrc[src * F + t];
        s += mv; s2 += mv * mv;
        mnv = fminf(mnv, mv); mxv = fmaxf(mxv, mv);
    }
    float mean = s / safe;
    float var  = fmaxf(s2 / safe - mean * mean, 0.f);
    meanO[o] = mean;
    stdO[o]  = sqrtf(var + 1e-5f);
    mnO[o]   = deg > 0 ? mnv : 0.f;
    mxO[o]   = deg > 0 ? mxv : 0.f;
}

// ----------------------- layer-2 post (N==1 output) -------------------------
__global__ __launch_bounds__(256) void k_post2(
    const float* __restrict__ h1, const float* __restrict__ mean,
    const float* __restrict__ mn, const float* __restrict__ mx,
    const float* __restrict__ stdv, const float* __restrict__ Wq,
    const float* __restrict__ bq, const float* __restrict__ Wl,
    const float* __restrict__ bl, const float* __restrict__ amp,
    const float* __restrict__ inva, float* __restrict__ h2) {
    __shared__ float red[256];
    const int n = blockIdx.x, t = threadIdx.x;
    const int o = n * 256 + t;
    float idp = h1[o]   * Wq[t]
              + mean[o] * Wq[256  + t] + mn[o]   * Wq[512  + t]
              + mx[o]   * Wq[768  + t] + stdv[o] * Wq[1024 + t];
    float ap  = mean[o] * Wq[1280 + t] + mn[o]   * Wq[1536 + t]
              + mx[o]   * Wq[1792 + t] + stdv[o] * Wq[2048 + t];
    float ip  = mean[o] * Wq[2304 + t] + mn[o]   * Wq[2560 + t]
              + mx[o]   * Wq[2816 + t] + stdv[o] * Wq[3072 + t];
    red[t] = idp + amp[n] * ap + inva[n] * ip;
    __syncthreads();
    for (int s = 128; s > 0; s >>= 1) { if (t < s) red[t] += red[t + s]; __syncthreads(); }
    if (t == 0) {
        float r = red[0] + bq[0];
        h2[n] = r * Wl[0] + bl[0];
    }
}

__global__ __launch_bounds__(256) void k_final(const float* __restrict__ h2,
                                               const float* __restrict__ Wout,
                                               const float* __restrict__ bout,
                                               float* __restrict__ out, int N) {
    __shared__ float red[256];
    int t = threadIdx.x; float s = 0.f;
    for (int n = t; n < N; n += 256) s += h2[n] * Wout[n];
    red[t] = s; __syncthreads();
    for (int st = 128; st > 0; st >>= 1) { if (t < st) red[t] += red[t + st]; __syncthreads(); }
    if (t == 0) out[0] = red[0] + bout[0];
}

// --------------------------------- launch ----------------------------------

extern "C" void kernel_launch(void* const* d_in, const int* in_sizes, int n_in,
                              void* d_out, int out_size, void* d_ws, size_t ws_size,
                              hipStream_t stream) {
    const float*     x       = (const float*)d_in[0];
    const long long* ei      = (const long long*)d_in[1];
    const float*     W_pre1  = (const float*)d_in[4];
    const float*     b_pre1  = (const float*)d_in[5];
    const float*     W_post1 = (const float*)d_in[6];
    const float*     b_post1 = (const float*)d_in[7];
    const float*     W_lin1  = (const float*)d_in[8];
    const float*     b_lin1  = (const float*)d_in[9];
    const float*     W_pre2  = (const float*)d_in[10];
    const float*     b_pre2  = (const float*)d_in[11];
    const float*     W_post2 = (const float*)d_in[12];
    const float*     b_post2 = (const float*)d_in[13];
    const float*     W_lin2  = (const float*)d_in[14];
    const float*     b_lin2  = (const float*)d_in[15];
    const float*     W_out   = (const float*)d_in[16];
    const float*     b_out   = (const float*)d_in[17];

    const int E  = in_sizes[2];          // 65536
    const int N  = in_sizes[16];         // 2048
    const int F1 = in_sizes[0] / N;      // 2048
    const int H  = in_sizes[7];          // 256

    uintptr_t basep = (uintptr_t)d_ws;
    auto carve = [&](size_t bytes) -> void* {
        void* p = (void*)basep;
        basep += (bytes + 255) & ~(size_t)255;
        return p;
    };
    // f32 buffers
    float* A1    = (float*)carve((size_t)N * F1 * 4);
    float* B1    = (float*)carve((size_t)N * F1 * 4);
    float* T     = (float*)carve((size_t)N * H * 4);
    float* h1    = (float*)carve((size_t)N * H * 4);
    float* A2    = (float*)carve((size_t)N * H * 4);
    float* B2    = (float*)carve((size_t)N * H * 4);
    float* mean2 = (float*)carve((size_t)N * H * 4);
    float* mn2   = (float*)carve((size_t)N * H * 4);
    float* mx2   = (float*)carve((size_t)N * H * 4);
    float* std2  = (float*)carve((size_t)N * H * 4);
    float* h2    = (float*)carve((size_t)N * 4);
    float* amp   = (float*)carve((size_t)N * 4);
    float* inva  = (float*)carve((size_t)N * 4);
    // bf16 operands
    __bf16* xb      = (__bf16*)carve((size_t)N * F1 * 2);
    __bf16* Wpre1b  = (__bf16*)carve((size_t)2 * F1 * F1 * 2);
    __bf16* Wpost1b = (__bf16*)carve((size_t)13 * F1 * H * 2);
    __bf16* Wlin1b  = (__bf16*)carve((size_t)H * H * 2);
    __bf16* Wpre2b  = (__bf16*)carve((size_t)2 * H * H * 2);
    __bf16* mean1b  = (__bf16*)carve((size_t)N * F1 * 2);
    __bf16* mn1b    = (__bf16*)carve((size_t)N * F1 * 2);
    __bf16* mx1b    = (__bf16*)carve((size_t)N * F1 * 2);
    __bf16* std1b   = (__bf16*)carve((size_t)N * F1 * 2);
    __bf16* Tb      = (__bf16*)carve((size_t)N * H * 2);
    __bf16* h1b     = (__bf16*)carve((size_t)N * H * 2);
    // graph
    int* deg     = (int*)carve((size_t)N * 4);
    int* rowptr  = (int*)carve((size_t)(N + 1) * 4);
    int* cursor  = (int*)carve((size_t)N * 4);
    int* ebuf    = (int*)carve((size_t)E * 4);

    const int nb = (N + 255) / 256, eb = (E + 255) / 256;
    auto cvt = [&](const float* in, __bf16* out, size_t n) {
        int n4 = (int)(n / 4);
        k_f32_to_bf16<<<(n4 + 255) / 256, 256, 0, stream>>>(in, out, n4);
    };

    // deterministic CSR + degree scalers
    k_zero_i32<<<nb, 256, 0, stream>>>(deg, N);
    k_zero_i32<<<nb, 256, 0, stream>>>(cursor, N);
    k_degree<<<eb, 256, 0, stream>>>(ei, deg, E);
    k_scan_amp<<<1, 256, 0, stream>>>(deg, rowptr, amp, inva, N);
    k_fill_csr<<<eb, 256, 0, stream>>>(ei, rowptr, cursor, ebuf, E);
    k_sort_buckets<<<nb, 256, 0, stream>>>(rowptr, ebuf, N);

    // one-time bf16 conversions of GEMM operands
    cvt(x,       xb,      (size_t)N * F1);
    cvt(W_pre1,  Wpre1b,  (size_t)2 * F1 * F1);
    cvt(W_post1, Wpost1b, (size_t)13 * F1 * H);
    cvt(W_lin1,  Wlin1b,  (size_t)H * H);
    cvt(W_pre2,  Wpre2b,  (size_t)2 * H * H);

    // ---- layer 1 ----
    dim3 gPre(F1 / 128, N / 64);
    k_gemm_bf16<false, false, true,  false><<<gPre, 256, 0, stream>>>(
        xb, Wpre1b, A1, F1, F1, F1, F1, b_pre1, nullptr);
    k_gemm_bf16<false, false, false, false><<<gPre, 256, 0, stream>>>(
        xb, Wpre1b + (size_t)F1 * F1, B1, F1, F1, F1, F1, nullptr, nullptr);

    k_edge_aggr_v<2><<<N, 256, 0, stream>>>(A1, B1, rowptr, ebuf,
                                            mean1b, mn1b, mx1b, std1b, F1);

    dim3 gH(H / 128, N / 64);
    const __bf16* aggrs[4] = {mean1b, mn1b, mx1b, std1b};
    k_gemm_bf16<false, false, true, false><<<gH, 256, 0, stream>>>(
        xb, Wpost1b, T, F1, F1, H, H, b_post1, nullptr);
    for (int i = 0; i < 4; i++)
        k_gemm_bf16<true, false, false, false><<<gH, 256, 0, stream>>>(
            aggrs[i], Wpost1b + (size_t)F1 * (1 + i) * H, T,
            F1, F1, H, H, nullptr, nullptr);
    for (int i = 0; i < 4; i++)
        k_gemm_bf16<true, false, false, true><<<gH, 256, 0, stream>>>(
            aggrs[i], Wpost1b + (size_t)F1 * (5 + i) * H, T,
            F1, F1, H, H, nullptr, amp);
    for (int i = 0; i < 4; i++)
        k_gemm_bf16<true, false, false, true><<<gH, 256, 0, stream>>>(
            aggrs[i], Wpost1b + (size_t)F1 * (9 + i) * H, T,
            F1, F1, H, H, nullptr, inva);

    cvt(T, Tb, (size_t)N * H);
    k_gemm_bf16<false, true, true, false><<<gH, 256, 0, stream>>>(
        Tb, Wlin1b, h1, H, H, H, H, b_lin1, nullptr);
    cvt(h1, h1b, (size_t)N * H);

    // ---- layer 2 ----
    k_gemm_bf16<false, false, true,  false><<<gH, 256, 0, stream>>>(
        h1b, Wpre2b, A2, H, H, H, H, b_pre2, nullptr);
    k_gemm_bf16<false, false, false, false><<<gH, 256, 0, stream>>>(
        h1b, Wpre2b + (size_t)H * H, B2, H, H, H, H, nullptr, nullptr);

    k_edge_aggr_s<<<N, 256, 0, stream>>>(A2, B2, rowptr, ebuf,
                                         mean2, mn2, mx2, std2, H);

    k_post2<<<N, 256, 0, stream>>>(h1, mean2, mn2, mx2, std2,
                                   W_post2, b_post2, W_lin2, b_lin2,
                                   amp, inva, h2);

    k_final<<<1, 256, 0, stream>>>(h2, W_out, b_out, (float*)d_out, N);
}
